// BaseLinearLayerWithLoRA_16157666967998
// MI455X (gfx1250) — matrix operand order repaired and checked by reading, not compiled, not run on hardware
//
#include <hip/hip_runtime.h>

// ---------------------------------------------------------------------------
// LoRA linear on gfx1250 (MI455X): base GEMM + punica add_lora as dense WMMA.
//
// Precision: fp32 inputs split into bf16 hi/lo planes on the fly; each
// 16x16x32 MAC block = 3x v_wmma_f32_16x16x32_bf16 (hi*hi + hi*lo + lo*hi),
// dropping only the lo*lo term (<= 2^-18 relative) -> fp32-class accuracy at
// 2.67x fewer matrix ops than the exact V_WMMA_F32_16X16X4_F32 path.
//
// Data movement: register-staged double-buffered LDS pipeline; global loads
// for chunk c+1 issue before the 24-WMMA burst for chunk c, so HBM latency
// overlaps matrix math. One workgroup barrier per K-chunk.
//
//   S      = mask(idx) .* (x @ Acat^T)        [T, 128]   (Acat = lora_a flat)
//   Btail  = transpose(lora_b)                [D_OUT,128]
//   out    = x @ W^T + S @ Btail^T + bias     [T, D_OUT]
// ---------------------------------------------------------------------------

typedef __attribute__((ext_vector_type(4)))  float   v4f;
typedef __attribute__((ext_vector_type(8)))  float   v8f;
typedef __attribute__((ext_vector_type(4)))  __bf16  v4bf;
typedef __attribute__((ext_vector_type(8)))  __bf16  v8bf;
typedef __attribute__((ext_vector_type(16))) __bf16  v16bf;

#define T_TOK  32768
#define DIN    2048
#define DOUT   2048
#define NLORA  8
#define RANK   16
#define CDIM   (NLORA * RANK)   // 128

#define BM 128        // block tile rows (tokens)
#define BN 128        // block tile cols
#define BK 32         // K chunk staged in LDS (one bf16-WMMA K depth)
#define LSTRIDE 40    // bf16 elements per padded LDS row (80B: conflict-free)
#define PLANE (BM * LSTRIDE)   // one bf16 plane (10240 B)
// LDS arena: [buffer 0/1][plane: Ahi,Alo,Bhi,Blo] -> 80 KB total
__device__ __forceinline__ __bf16* plane(__bf16* arena, int p, int q) {
  return arena + (p * 4 + q) * PLANE;
}

// Split one v4f into bf16 hi/lo planes (round-to-nearest-even each step).
__device__ __forceinline__ void split4(v4f v, v4bf& hi, v4bf& lo) {
#pragma unroll
  for (int e = 0; e < 4; ++e) {
    __bf16 h = (__bf16)v[e];
    hi[e] = h;
    lo[e] = (__bf16)(v[e] - (float)h);
  }
}

// Assemble a 16-element bf16 fragment from two 16B LDS runs (2x ds_load_b128).
__device__ __forceinline__ v16bf frag2(const __bf16* p0, const __bf16* p1) {
  v8bf a = *(const v8bf*)p0;
  v8bf b = *(const v8bf*)p1;
  return __builtin_shufflevector(a, b, 0, 1, 2, 3, 4, 5, 6, 7,
                                 8, 9, 10, 11, 12, 13, 14, 15);
}

// Cooperative staging: 256 threads cover a 128x32 fp32 tile; thread covers
// rows r0, r0+32, r0+64, r0+96 at fixed 4-float column c4.
__device__ __forceinline__ void stage_load(const float* __restrict__ Row,
                                           long stride, int k0,
                                           int r0, int c4, v4f out[4]) {
#pragma unroll
  for (int i = 0; i < 4; ++i)
    out[i] = *(const v4f*)(Row + (long)(r0 + i * 32) * stride + k0 + c4);
}

__device__ __forceinline__ void stage_store(__bf16* hiP, __bf16* loP,
                                            int r0, int c4, const v4f in[4]) {
#pragma unroll
  for (int i = 0; i < 4; ++i) {
    v4bf h, l;
    split4(in[i], h, l);
    *(v4bf*)(hiP + (r0 + i * 32) * LSTRIDE + c4) = h;
    *(v4bf*)(loP + (r0 + i * 32) * LSTRIDE + c4) = l;
  }
}

// ---------------------------------------------------------------------------
// Double-buffered GEMM inner loop: acc[i][j] += A(m,k)*B(n,k), k in [0,K).
//   A element (m,k) = Arow[m*a_stride + k], m in [0,BM)
//   B element (n,k) = Brow[n*b_stride + k], n in [0,BN)
// 8 waves in 4(M) x 2(N); each wave owns 32(M) x 64(N): 2x4 accum tiles,
// 24 ds_load_b128 per 24 WMMAs per chunk.
// ---------------------------------------------------------------------------
__device__ __forceinline__ void gemm_tile_loop(
    const float* __restrict__ Arow, long a_stride,
    const float* __restrict__ Brow, long b_stride,
    int K, __bf16* arena, v8f acc[2][4])
{
  const int tid  = threadIdx.x;
  const int wave = tid >> 5;      // 0..7 (wave32)
  const int lane = tid & 31;
  const int half = lane >> 4;
  const int l16  = lane & 15;
  const int wm   = wave & 3;      // M sub-block 0..3  (rows wm*32 ..)
  const int wn   = wave >> 2;     // N sub-block 0..1  (cols wn*64 ..)
  const int r0   = tid >> 3;      // staging base row 0..31
  const int c4   = (tid & 7) << 2;

  v4f ra[4], rb[4];

  // Prologue: fill buffer 0 with chunk 0.
  stage_load(Arow, a_stride, 0, r0, c4, ra);
  stage_load(Brow, b_stride, 0, r0, c4, rb);
  stage_store(plane(arena, 0, 0), plane(arena, 0, 1), r0, c4, ra);
  stage_store(plane(arena, 0, 2), plane(arena, 0, 3), r0, c4, rb);
  __syncthreads();

  const int nch = K / BK;
  for (int c = 0; c < nch; ++c) {
    const int p = c & 1;
    const bool more = (c + 1 < nch);

    // Issue next chunk's global loads first: latency hides under the WMMAs.
    if (more) {
      stage_load(Arow, a_stride, (c + 1) * BK, r0, c4, ra);
      stage_load(Brow, b_stride, (c + 1) * BK, r0, c4, rb);
    }
    if (c + 2 < nch) {
      // Warm L2 for chunk c+2 (global_prefetch_b8 on gfx1250).
      __builtin_prefetch(Arow + (long)(tid >> 1) * a_stride + (c + 2) * BK, 0, 0);
      __builtin_prefetch(Brow + (long)(tid >> 1) * b_stride + (c + 2) * BK, 0, 0);
    }

    __bf16* lAhi = plane(arena, p, 0);
    __bf16* lAlo = plane(arena, p, 1);
    __bf16* lBhi = plane(arena, p, 2);
    __bf16* lBlo = plane(arena, p, 3);

    // A fragments: 16-bit A 16x32 layout — lane l16 = M; per lane two runs:
    //   elements 0..7  = K [half*8, +8), elements 8..15 = K [16+half*8, +8)
    v16bf ah[2], al[2];
#pragma unroll
    for (int i = 0; i < 2; ++i) {
      const int ra_off = (wm * 32 + i * 16 + l16) * LSTRIDE + half * 8;
      ah[i] = frag2(lAhi + ra_off, lAhi + ra_off + 16);
      al[i] = frag2(lAlo + ra_off, lAlo + ra_off + 16);
    }
    // B fragments: 32x16 layout — lane l16 = N; elements 0..7 =
    //   K [half*8, +8), elements 8..15 = K [16+half*8, +8) (two 16B runs).
#pragma unroll
    for (int j = 0; j < 4; ++j) {
      const int rb_off = (wn * 64 + j * 16 + l16) * LSTRIDE + half * 8;
      v16bf bh = frag2(lBhi + rb_off, lBhi + rb_off + 16);
      v16bf bl = frag2(lBlo + rb_off, lBlo + rb_off + 16);
#pragma unroll
      for (int i = 0; i < 2; ++i) {
        acc[i][j] = __builtin_amdgcn_wmma_f32_16x16x32_bf16(
            false, ah[i], false, bh, (short)0, acc[i][j], false, false);
        acc[i][j] = __builtin_amdgcn_wmma_f32_16x16x32_bf16(
            false, ah[i], false, bl, (short)0, acc[i][j], false, false);
        acc[i][j] = __builtin_amdgcn_wmma_f32_16x16x32_bf16(
            false, al[i], false, bh, (short)0, acc[i][j], false, false);
      }
    }

    // Convert + store next chunk into the other buffer, then one barrier.
    if (more) {
      stage_store(plane(arena, p ^ 1, 0), plane(arena, p ^ 1, 1), r0, c4, ra);
      stage_store(plane(arena, p ^ 1, 2), plane(arena, p ^ 1, 3), r0, c4, rb);
    }
    __syncthreads();
  }
}

// ---------------------------------------------------------------------------
// Kernel 1: Btail[o*128 + c] = lora_b[c/16, 0, o, c%16]
// ---------------------------------------------------------------------------
__global__ __launch_bounds__(256) void btail_kernel(
    const float* __restrict__ lora_b, float* __restrict__ Btail)
{
  int i = blockIdx.x * 256 + threadIdx.x;
  if (i < DOUT * CDIM) {
    int o = i >> 7;
    int cc = i & (CDIM - 1);
    int l = cc >> 4;
    int r = cc & (RANK - 1);
    Btail[i] = lora_b[(long)l * DOUT * RANK + (long)o * RANK + r];
  }
}

// ---------------------------------------------------------------------------
// Kernel 2: masked shrink  S = mask .* (x @ Acat^T)      [T, 128]
// ---------------------------------------------------------------------------
__global__ __launch_bounds__(256) void shrink_kernel(
    const float* __restrict__ x, const float* __restrict__ lora_a,
    const int* __restrict__ idx, float* __restrict__ S)
{
  __shared__ __bf16 arena[2 * 4 * PLANE];
  v8f acc[2][4] = {};

  const long mblk = (long)blockIdx.x * BM;
  gemm_tile_loop(x + mblk * DIN, DIN, lora_a, DIN, DIN, arena, acc);

  const int tid = threadIdx.x, wave = tid >> 5, lane = tid & 31;
  const int half = lane >> 4, l16 = lane & 15;
  const int wm = wave & 3, wn = wave >> 2;
  // C/D layout: VGPR v -> row (v + half*8), col l16 within each 16x16 tile.
#pragma unroll
  for (int i = 0; i < 2; ++i) {
#pragma unroll
    for (int v = 0; v < 8; ++v) {
      long t = mblk + wm * 32 + i * 16 + half * 8 + v;
      int lad = idx[t];
#pragma unroll
      for (int j = 0; j < 4; ++j) {
        int cc = wn * 64 + j * 16 + l16;
        S[t * CDIM + cc] = (lad == (cc >> 4)) ? acc[i][j][v] : 0.0f;
      }
    }
  }
}

// ---------------------------------------------------------------------------
// Kernel 3: out = x @ W^T + S @ Btail^T + bias
// ---------------------------------------------------------------------------
__global__ __launch_bounds__(256) void lora_main_kernel(
    const float* __restrict__ x, const float* __restrict__ W,
    const float* __restrict__ bias, const float* __restrict__ S,
    const float* __restrict__ Btail, float* __restrict__ out)
{
  __shared__ __bf16 arena[2 * 4 * PLANE];
  v8f acc[2][4] = {};

  const long mblk = (long)blockIdx.x * BM;
  const long nblk = (long)blockIdx.y * BN;

  // Base GEMM over K = 2048, then LoRA tail over K = 128 into same accums.
  gemm_tile_loop(x + mblk * DIN, DIN, W + nblk * DIN, DIN, DIN, arena, acc);
  gemm_tile_loop(S + mblk * CDIM, CDIM, Btail + nblk * CDIM, CDIM, CDIM,
                 arena, acc);

  const int tid = threadIdx.x, wave = tid >> 5, lane = tid & 31;
  const int half = lane >> 4, l16 = lane & 15;
  const int wm = wave & 3, wn = wave >> 2;
#pragma unroll
  for (int i = 0; i < 2; ++i) {
#pragma unroll
    for (int v = 0; v < 8; ++v) {
      long t = mblk + wm * 32 + i * 16 + half * 8 + v;
#pragma unroll
      for (int j = 0; j < 4; ++j) {
        int o = (int)nblk + wn * 64 + j * 16 + l16;
        out[t * DOUT + o] = acc[i][j][v] + bias[o];
      }
    }
  }
}

// ---------------------------------------------------------------------------
extern "C" void kernel_launch(void* const* d_in, const int* in_sizes, int n_in,
                              void* d_out, int out_size, void* d_ws, size_t ws_size,
                              hipStream_t stream)
{
  const float* x      = (const float*)d_in[0];  // [T, DIN]
  const float* W      = (const float*)d_in[1];  // [DOUT, DIN]
  const float* bias   = (const float*)d_in[2];  // [DOUT]
  const float* lora_a = (const float*)d_in[3];  // [8,1,16,DIN] == [128, DIN]
  const float* lora_b = (const float*)d_in[4];  // [8,1,DOUT,16]
  const int*   idx    = (const int*)d_in[5];    // [T]
  float* out = (float*)d_out;

  float* S     = (float*)d_ws;                   // [T, 128]    (16 MB)
  float* Btail = S + (size_t)T_TOK * CDIM;       // [DOUT, 128] (1 MB)

  btail_kernel<<<(DOUT * CDIM + 255) / 256, 256, 0, stream>>>(lora_b, Btail);
  shrink_kernel<<<T_TOK / BM, 256, 0, stream>>>(x, lora_a, idx, S);
  lora_main_kernel<<<dim3(T_TOK / BM, DOUT / BN), 256, 0, stream>>>(
      x, W, bias, S, Btail, out);
}
